// LSTMModel_89515708383705
// MI455X (gfx1250) — matrix operand order repaired and checked by reading, not compiled, not run on hardware
//
#include <hip/hip_runtime.h>
#include <hip/hip_bf16.h>

typedef __bf16 bf16;
typedef __attribute__((ext_vector_type(16))) __bf16 v16bf;
typedef __attribute__((ext_vector_type(8)))  float  v8f;

static constexpr int BATCH = 256;
static constexpr int SEQ   = 336;
static constexpr int DIN   = 128;
static constexpr int HID   = 1024;
static constexpr int NG    = 4 * HID;   // 4096 gate width
static constexpr int NLAB  = 24;

static constexpr int BM = 64;
static constexpr int BN = 128;
static constexpr int BK = 32;
static constexpr int LDP = BK + 8;      // padded LDS row (80B, 16B-aligned)

// ---- CDNA5 async direct-to-LDS copy (ASYNCcnt-tracked, bypasses VGPRs) ----
__device__ __forceinline__ void async_ld_b128(unsigned lds_off, const void* gaddr) {
  asm volatile("global_load_async_to_lds_b128 %0, %1, off"
               :: "v"(lds_off), "v"(gaddr) : "memory");
}
__device__ __forceinline__ void wait_async0() {
  asm volatile("s_wait_asynccnt 0x0" ::: "memory");
}

// ---------------- WMMA GEMM: Z = [A1 | A2] @ Bt^T + bias ----------------
// A1: [BATCH][K1] rows strided lda1 (bf16), A2: [BATCH][K2] strided lda2 (bf16)
// Bt: [NG][K1+K2] bf16 (K contiguous),  Z: [BATCH][NG] f32
// K1, K2 are multiples of BK. Double-buffered, async-to-LDS pipeline.
__global__ __launch_bounds__(256) void gemm_lstm(
    const bf16* __restrict__ A1, int lda1, int K1,
    const bf16* __restrict__ A2, int lda2, int K2,
    const bf16* __restrict__ Bt,
    const float* __restrict__ bias,
    float* __restrict__ Z)
{
  const int Ktot = K1 + K2;
  const int nk   = Ktot / BK;
  __shared__ bf16 As[2][BM][LDP];
  __shared__ bf16 Bs[2][BN][LDP];

  const int tid  = threadIdx.x;
  const int lane = tid & 31;
  const int wave = tid >> 5;
  const int wm   = wave & 1;        // 2 waves along M
  const int wn   = wave >> 1;       // 4 waves along N
  const int l15  = lane & 15;
  const int half = lane >> 4;       // lane half selects K-chunk / M-half
  const int blockN0 = blockIdx.x * BN;
  const int blockM0 = blockIdx.y * BM;

  // init accumulators with bias[n] (C layout: lane's N = l15 for both halves)
  v8f acc[2][2];
  #pragma unroll
  for (int mi = 0; mi < 2; ++mi)
    #pragma unroll
    for (int ni = 0; ni < 2; ++ni) {
      float bv = bias[blockN0 + wn * 32 + ni * 16 + l15];
      acc[mi][ni] = (v8f){bv, bv, bv, bv, bv, bv, bv, bv};
    }

  // --- per-thread staging coordinates / precomputed pointers ---
  const int arow = tid >> 2;             // 0..63
  const int acol = (tid & 3) * 8;        // 16B chunk within K-tile
  const int bnr  = tid >> 2;             // 0..63 (chunk 0); chunk 1 adds +64 rows
  const bf16* pA1 = A1 + (size_t)(blockM0 + arow) * lda1 + acol;
  const bf16* pA2 = A2 + (size_t)(blockM0 + arow) * lda2 + acol;
  const bf16* pB0 = Bt + (size_t)(blockN0 + bnr) * Ktot + acol;
  const bf16* pB1 = pB0 + (size_t)64 * Ktot;

  // LDS byte addresses (generic-pointer low 32 bits == LDS offset)
  unsigned ldsA[2], ldsB0[2], ldsB1[2];
  #pragma unroll
  for (int b = 0; b < 2; ++b) {
    ldsA[b]  = (unsigned)(uintptr_t)&As[b][arow][acol];
    ldsB0[b] = (unsigned)(uintptr_t)&Bs[b][bnr][acol];
    ldsB1[b] = (unsigned)(uintptr_t)&Bs[b][bnr + 64][acol];
  }

  auto async_fetch = [&](int k0, int buf) {
    const bf16* ga = (k0 < K1) ? (pA1 + k0) : (pA2 + (k0 - K1));
    async_ld_b128(ldsA[buf],  ga);
    async_ld_b128(ldsB0[buf], pB0 + k0);
    async_ld_b128(ldsB1[buf], pB1 + k0);
  };

  async_fetch(0, 0);
  wait_async0();
  __syncthreads();

  for (int i = 0; i < nk; ++i) {
    const int cur = i & 1;
    const bool more = (i + 1) < nk;
    if (more) async_fetch((i + 1) * BK, cur ^ 1);  // fills back buffer during WMMA

    // --- fragments per ISA 16-bit layouts ---
    v16bf afrag[2], bfrag[2];
    #pragma unroll
    for (int mi = 0; mi < 2; ++mi) {
      int r  = wm * 32 + mi * 16 + l15;  // A: lane = M row
      int kb = half * 8;                 // lanes0-15: K 0-7 & 16-23; lanes16-31: +8
      uint4* d = (uint4*)&afrag[mi];
      d[0] = *(const uint4*)&As[cur][r][kb];
      d[1] = *(const uint4*)&As[cur][r][kb + 16];
    }
    #pragma unroll
    for (int ni = 0; ni < 2; ++ni) {
      int nc = wn * 32 + ni * 16 + l15;  // B: lane = N col
      int ks = half * 8;                 // lanes0-15: K 0-7 & 16-23; lanes16-31: +8
      uint4* d = (uint4*)&bfrag[ni];
      d[0] = *(const uint4*)&Bs[cur][nc][ks];
      d[1] = *(const uint4*)&Bs[cur][nc][ks + 16];
    }

    #pragma unroll
    for (int mi = 0; mi < 2; ++mi)
      #pragma unroll
      for (int ni = 0; ni < 2; ++ni)
        acc[mi][ni] = __builtin_amdgcn_wmma_f32_16x16x32_bf16(
            false, afrag[mi], false, bfrag[ni],
            (short)0, acc[mi][ni], false, false);

    if (more) wait_async0();             // back buffer complete before barrier
    __syncthreads();
  }

  // --- store: VGPR r -> M = base + half*8 + r, N = base + l15 ---
  #pragma unroll
  for (int mi = 0; mi < 2; ++mi) {
    int m0 = blockM0 + wm * 32 + mi * 16 + half * 8;
    #pragma unroll
    for (int ni = 0; ni < 2; ++ni) {
      int n = blockN0 + wn * 32 + ni * 16 + l15;
      float* zp = Z + (size_t)m0 * NG + n;
      #pragma unroll
      for (int r = 0; r < 8; ++r)
        zp[(size_t)r * NG] = acc[mi][ni][r];
    }
  }
}

// ---------------- LSTM cell elementwise ----------------
__device__ __forceinline__ float sigm(float x) { return 1.0f / (1.0f + __expf(-x)); }

__global__ __launch_bounds__(256) void lstm_cell(
    const float* __restrict__ Z, float* __restrict__ c,
    bf16* __restrict__ h, float forget_bias)
{
  int idx = blockIdx.x * 256 + threadIdx.x;   // b*HID + hh
  int b  = idx >> 10;
  int hh = idx & (HID - 1);
  const float* zr = Z + (size_t)b * NG;
  float zi = zr[hh];
  float zj = zr[HID + hh];
  float zf = zr[2 * HID + hh];
  float zo = zr[3 * HID + hh];
  float cv = c[idx];
  float nc = cv * sigm(zf + forget_bias) + sigm(zi) * tanhf(zj);
  float nh = tanhf(nc) * sigm(zo);
  c[idx] = nc;
  h[idx] = (bf16)nh;
}

// ---------------- dense head + MSE loss ----------------
__global__ void zero_out(float* out) { out[0] = 0.0f; }

__global__ __launch_bounds__(256) void head_loss(
    const bf16* __restrict__ h2, const float* __restrict__ Wd,
    const float* __restrict__ bd, const float* __restrict__ labels,
    float* __restrict__ out)
{
  int n = blockIdx.x;      // 0..23
  int b = threadIdx.x;     // 0..255
  const bf16* hp = h2 + (size_t)b * HID;
  float s = 0.0f;
  for (int k = 0; k < HID; ++k)
    s += (float)hp[k] * Wd[k * NLAB + n];
  float d = s + bd[n] - labels[b * NLAB + n];
  __shared__ float red[256];
  red[b] = d * d;
  __syncthreads();
  for (int off = 128; off > 0; off >>= 1) {
    if (b < off) red[b] += red[b + off];
    __syncthreads();
  }
  if (b == 0) atomicAdd(out, red[0] * (1.0f / (BATCH * NLAB)));
}

// ---------------- setup kernels ----------------
__global__ void cvt_f32_bf16(const float* __restrict__ in, bf16* __restrict__ o, int n) {
  int i = blockIdx.x * 256 + threadIdx.x;
  if (i < n) o[i] = (bf16)in[i];
}
__global__ void transpose_cvt(const float* __restrict__ in, bf16* __restrict__ o, int K, int N) {
  // o[n*K + k] = in[k*N + n]
  int i = blockIdx.x * 256 + threadIdx.x;
  if (i < K * N) {
    int k = i % K;
    int n = i / K;
    o[i] = (bf16)in[(size_t)k * N + n];
  }
}
__global__ void zero_f32(float* p, int n)  { int i = blockIdx.x*256+threadIdx.x; if (i < n) p[i] = 0.0f; }
__global__ void zero_bf16(bf16* p, int n)  { int i = blockIdx.x*256+threadIdx.x; if (i < n) p[i] = (bf16)0.0f; }

extern "C" void kernel_launch(void* const* d_in, const int* in_sizes, int n_in,
                              void* d_out, int out_size, void* d_ws, size_t ws_size,
                              hipStream_t stream) {
  const float* features = (const float*)d_in[0];
  const float* labels   = (const float*)d_in[1];
  const float* W1 = (const float*)d_in[2];
  const float* b1 = (const float*)d_in[3];
  const float* W2 = (const float*)d_in[4];
  const float* b2 = (const float*)d_in[5];
  const float* Wd = (const float*)d_in[6];
  const float* bd = (const float*)d_in[7];
  float* out = (float*)d_out;

  // workspace carve (256B aligned)
  char* ws = (char*)d_ws;
  size_t off = 0;
  auto carve = [&](size_t bytes) -> void* {
    void* p = ws + off;
    off = (off + bytes + 255) & ~(size_t)255;
    return p;
  };
  bf16*  Xbf = (bf16*)carve((size_t)BATCH * SEQ * DIN * 2);
  bf16*  W1t = (bf16*)carve((size_t)NG * (DIN + HID) * 2);
  bf16*  W2t = (bf16*)carve((size_t)NG * (2 * HID) * 2);
  float* Zb  = (float*)carve((size_t)BATCH * NG * 4);
  float* c1  = (float*)carve((size_t)BATCH * HID * 4);
  float* c2  = (float*)carve((size_t)BATCH * HID * 4);
  bf16*  h1  = (bf16*)carve((size_t)BATCH * HID * 2);
  bf16*  h2  = (bf16*)carve((size_t)BATCH * HID * 2);

  // one-time conversions (weights resident as bf16 [N][K] -> L2-friendly GEMM B)
  int nf = BATCH * SEQ * DIN;
  cvt_f32_bf16<<<(nf + 255) / 256, 256, 0, stream>>>(features, Xbf, nf);
  transpose_cvt<<<(NG * (DIN + HID) + 255) / 256, 256, 0, stream>>>(W1, W1t, DIN + HID, NG);
  transpose_cvt<<<(NG * 2 * HID + 255) / 256, 256, 0, stream>>>(W2, W2t, 2 * HID, NG);
  int ns = BATCH * HID;
  zero_f32 <<<(ns + 255) / 256, 256, 0, stream>>>(c1, ns);
  zero_f32 <<<(ns + 255) / 256, 256, 0, stream>>>(c2, ns);
  zero_bf16<<<(ns + 255) / 256, 256, 0, stream>>>(h1, ns);
  zero_bf16<<<(ns + 255) / 256, 256, 0, stream>>>(h2, ns);

  dim3 ggrid(NG / BN, BATCH / BM);   // 32 x 4 = 128 workgroups
  for (int t = 0; t < SEQ; ++t) {
    // layer 1: [x_t | h1] @ W1 + b1
    gemm_lstm<<<ggrid, 256, 0, stream>>>(Xbf + (size_t)t * DIN, SEQ * DIN, DIN,
                                         h1, HID, HID, W1t, b1, Zb);
    lstm_cell<<<ns / 256, 256, 0, stream>>>(Zb, c1, h1, 1.0f);
    // layer 2: [h1 | h2] @ W2 + b2
    gemm_lstm<<<ggrid, 256, 0, stream>>>(h1, HID, HID,
                                         h2, HID, HID, W2t, b2, Zb);
    lstm_cell<<<ns / 256, 256, 0, stream>>>(Zb, c2, h2, 1.0f);
  }

  zero_out<<<1, 1, 0, stream>>>(out);
  head_loss<<<NLAB, 256, 0, stream>>>(h2, Wd, bd, labels, out);
}
